// WindowAttention_16080357556464
// MI455X (gfx1250) — matrix operand order repaired and checked by reading, not compiled, not run on hardware
//
#include <hip/hip_runtime.h>
#include <hip/hip_bf16.h>
#include <stdint.h>

// ---------------- problem constants ----------------
#define DIM      512
#define NTOK     64
#define HEADS    16
#define HD       32
#define NWIN     1024
#define NWINDOWS 2048
#define THREADS  512
#define WAVES    16

typedef __bf16 bf16_t;
typedef __attribute__((ext_vector_type(16))) __bf16 v16bf;
typedef __attribute__((ext_vector_type(8)))  float  v8f;

union FragBF { v16bf v; uint4 u[2]; };
union AccF   { v8f   v; float f[8]; };

__device__ __forceinline__ v8f zero8() {
  v8f z = {0.f, 0.f, 0.f, 0.f, 0.f, 0.f, 0.f, 0.f};
  return z;
}

// padded LDS row strides (in bf16 elements); keep 16B alignment of fragments
#define XS  520   // x / Q / K / AO rows (512 + 8 pad) -> 1040 B (65*16)
#define VTS 72    // V^T rows: 64 tokens + 8 pad -> 144 B (9*16)

// ---------------------------------------------------------------
// Prep: f32 weights -> bf16 in workspace; expand relative-position
// bias to [H,64,64] f32 so the hot kernel does flat loads.
// ---------------------------------------------------------------
__global__ void winattn_prep(const float* __restrict__ qkv_w,
                             const float* __restrict__ proj_w,
                             const float* __restrict__ bias_table,
                             const int*   __restrict__ rel_idx,
                             bf16_t* __restrict__ qkv_wb,
                             bf16_t* __restrict__ proj_wb,
                             float*  __restrict__ bias_exp) {
  int i = blockIdx.x * blockDim.x + threadIdx.x;
  if (i < 3 * DIM * DIM) qkv_wb[i] = (bf16_t)qkv_w[i];
  if (i < DIM * DIM)     proj_wb[i] = (bf16_t)proj_w[i];
  if (i < HEADS * NTOK * NTOK) {
    int h = i >> 12;          // / 4096
    int mn = i & 4095;
    bias_exp[i] = bias_table[rel_idx[mn] * HEADS + h];
  }
}

// ---------------------------------------------------------------
// Fused window attention: one block = one window, 16 waves.
// ---------------------------------------------------------------
__global__ __launch_bounds__(THREADS, 1)
void winattn_main(const float*  __restrict__ x,
                  const float*  __restrict__ mask,
                  const bf16_t* __restrict__ qkv_wb,
                  const float*  __restrict__ qkv_b,
                  const bf16_t* __restrict__ proj_wb,
                  const float*  __restrict__ proj_b,
                  const float*  __restrict__ bias_exp,
                  float*        __restrict__ out) {
  __shared__ bf16_t sX [NTOK * XS];     // staged x; reused as attn-out (AO)
  __shared__ bf16_t sQ [NTOK * XS];
  __shared__ bf16_t sK [NTOK * XS];
  __shared__ bf16_t sVT[DIM  * VTS];    // V transposed: [channel][token]
  __shared__ bf16_t sP [WAVES][16 * 64];// per-wave softmax-prob transpose pad
  __shared__ float  sMask[NTOK * NTOK]; // async-staged window mask (16 KB)

  const int win  = blockIdx.x;
  const int tid  = threadIdx.x;
  const int wave = tid >> 5;
  const int lane = tid & 31;
  const int lr   = lane & 15;   // row (A) / col (B,C) selector
  const int lh   = lane >> 4;   // half selector

  // ---- async-stage window mask into LDS (overlaps QKV GEMM) ----
  {
    uint64_t mbase = (uint64_t)(uintptr_t)(mask + (size_t)(win % NWIN) * NTOK * NTOK);
    uint32_t lbase = (uint32_t)(uintptr_t)(void*)sMask;
    for (int c = tid; c < NTOK * NTOK / 4; c += THREADS) {  // 16B chunks
      uint32_t loff = lbase + (uint32_t)c * 16u;
      uint32_t goff = (uint32_t)c * 16u;
      asm volatile("global_load_async_to_lds_b128 %0, %1, %2"
                   :: "v"(loff), "v"(goff), "s"(mbase) : "memory");
    }
  }

  // ---------------- stage x -> LDS as bf16 ----------------
  {
    const float4* xg = (const float4*)(x + (size_t)win * NTOK * DIM);
    for (int i = tid; i < NTOK * DIM / 4; i += THREADS) {
      float4 f = xg[i];
      int row = i >> 7;            // (i*4)/512
      int col = (i & 127) << 2;    // (i*4)%512
      bf16_t* d = &sX[row * XS + col];
      d[0] = (bf16_t)f.x; d[1] = (bf16_t)f.y;
      d[2] = (bf16_t)f.z; d[3] = (bf16_t)f.w;
    }
  }
  __syncthreads();

  // ---------------- QKV GEMM: [64,512] @ W^T -> [64,1536] ----------------
  // wave owns output columns [wave*96, wave*96+96): 3 pairs of 16-col tiles
  for (int np = 0; np < 3; ++np) {
    const int o0  = wave * 96 + np * 32;
    const int oc0 = o0 + lr;
    const int oc1 = o0 + 16 + lr;
    const bf16_t* wrow0 = qkv_wb + (size_t)oc0 * DIM;
    const bf16_t* wrow1 = qkv_wb + (size_t)oc1 * DIM;
    AccF acc0[4], acc1[4];
#pragma unroll
    for (int m = 0; m < 4; ++m) { acc0[m].v = zero8(); acc1[m].v = zero8(); }

    for (int kk = 0; kk < DIM; kk += 32) {
      __builtin_prefetch(wrow0 + kk + 128, 0, 1);     // global_prefetch_b8
      __builtin_prefetch(wrow1 + kk + 128, 0, 1);
      FragBF bf0, bf1;                                // B: K at h*8, 16+h*8
      bf0.u[0] = *(const uint4*)(wrow0 + kk + lh * 8);
      bf0.u[1] = *(const uint4*)(wrow0 + kk + 16 + lh * 8);
      bf1.u[0] = *(const uint4*)(wrow1 + kk + lh * 8);
      bf1.u[1] = *(const uint4*)(wrow1 + kk + 16 + lh * 8);
#pragma unroll
      for (int m = 0; m < 4; ++m) {
        FragBF afrag;                                 // A: K at h*8, 16+h*8
        const bf16_t* ar = &sX[(m * 16 + lr) * XS + kk];
        afrag.u[0] = *(const uint4*)(ar + lh * 8);
        afrag.u[1] = *(const uint4*)(ar + 16 + lh * 8);
        acc0[m].v = __builtin_amdgcn_wmma_f32_16x16x32_bf16(
            false, afrag.v, false, bf0.v, (short)0, acc0[m].v, false, false);
        acc1[m].v = __builtin_amdgcn_wmma_f32_16x16x32_bf16(
            false, afrag.v, false, bf1.v, (short)0, acc1[m].v, false, false);
      }
    }
    const float b0 = qkv_b[oc0];
    const float b1 = qkv_b[oc1];
#pragma unroll
    for (int m = 0; m < 4; ++m) {
#pragma unroll
      for (int i = 0; i < 8; ++i) {
        int row = m * 16 + i + lh * 8;                // token index
        float v0 = acc0[m].f[i] + b0;
        float v1 = acc1[m].f[i] + b1;
        if (oc0 < DIM)          sQ[row * XS + oc0]               = (bf16_t)v0;
        else if (oc0 < 2 * DIM) sK[row * XS + (oc0 - DIM)]       = (bf16_t)v0;
        else                    sVT[(oc0 - 2 * DIM) * VTS + row] = (bf16_t)v0;
        if (oc1 < DIM)          sQ[row * XS + oc1]               = (bf16_t)v1;
        else if (oc1 < 2 * DIM) sK[row * XS + (oc1 - DIM)]       = (bf16_t)v1;
        else                    sVT[(oc1 - 2 * DIM) * VTS + row] = (bf16_t)v1;
      }
    }
  }
  asm volatile("s_wait_asynccnt 0" ::: "memory");     // mask copy landed
  __syncthreads();

  // ---------------- attention: one head per wave ----------------
  const int   h      = wave;
  const float scale  = 0.17677669529663687f;          // 1/sqrt(32)
  const float* biash = bias_exp + (size_t)h * NTOK * NTOK;
  bf16_t* Pbuf = sP[wave];

  for (int mt = 0; mt < 4; ++mt) {
    // S[16,64] = q_h[mt] @ k_h^T  (K = 32 = hd, one WMMA per n-tile)
    FragBF aq;
    const bf16_t* qr = &sQ[(mt * 16 + lr) * XS + h * HD];
    aq.u[0] = *(const uint4*)(qr + lh * 8);
    aq.u[1] = *(const uint4*)(qr + 16 + lh * 8);

    AccF s[4];
#pragma unroll
    for (int nt = 0; nt < 4; ++nt) {
      FragBF bk;                                      // B: col = key token
      const bf16_t* kr = &sK[(nt * 16 + lr) * XS + h * HD];
      bk.u[0] = *(const uint4*)(kr + lh * 8);
      bk.u[1] = *(const uint4*)(kr + 16 + lh * 8);
      s[nt].v = __builtin_amdgcn_wmma_f32_16x16x32_bf16(
          false, aq.v, false, bk.v, (short)0, zero8(), false, false);
    }

    // scale + relative-position bias + window mask (mask from LDS)
#pragma unroll
    for (int nt = 0; nt < 4; ++nt) {
#pragma unroll
      for (int i = 0; i < 8; ++i) {
        int m = mt * 16 + i + lh * 8;
        int n = nt * 16 + lr;
        s[nt].f[i] = s[nt].f[i] * scale + biash[m * 64 + n] + sMask[m * 64 + n];
      }
    }

    // row softmax: reduce across 4 n-tiles and across 16-lane half
#pragma unroll
    for (int i = 0; i < 8; ++i) {
      float mx = fmaxf(fmaxf(s[0].f[i], s[1].f[i]),
                       fmaxf(s[2].f[i], s[3].f[i]));
      for (int off = 1; off < 16; off <<= 1)
        mx = fmaxf(mx, __shfl_xor(mx, off, 32));
      float sum = 0.f;
#pragma unroll
      for (int nt = 0; nt < 4; ++nt) {
        float e = __expf(s[nt].f[i] - mx);
        s[nt].f[i] = e;
        sum += e;
      }
      for (int off = 1; off < 16; off <<= 1)
        sum += __shfl_xor(sum, off, 32);
      float inv = 1.f / sum;
#pragma unroll
      for (int nt = 0; nt < 4; ++nt) s[nt].f[i] *= inv;
    }

    // transpose P (C-layout -> row-major) through per-wave LDS scratch
#pragma unroll
    for (int nt = 0; nt < 4; ++nt) {
#pragma unroll
      for (int i = 0; i < 8; ++i) {
        int r = i + lh * 8;
        int n = nt * 16 + lr;
        Pbuf[r * 64 + n] = (bf16_t)s[nt].f[i];
      }
    }
    asm volatile("s_wait_dscnt 0" ::: "memory");      // intra-wave LDS RAW

    // O[16,32] = P @ v_h   (K = 64 tokens, 2 steps; 2 hd n-tiles)
    AccF o[2];
    o[0].v = zero8(); o[1].v = zero8();
#pragma unroll
    for (int ks = 0; ks < 2; ++ks) {
      FragBF ap;
      const bf16_t* pr = &Pbuf[lr * 64 + ks * 32];
      ap.u[0] = *(const uint4*)(pr + lh * 8);
      ap.u[1] = *(const uint4*)(pr + 16 + lh * 8);
#pragma unroll
      for (int nt = 0; nt < 2; ++nt) {
        FragBF bv;                                    // B from V^T rows
        const bf16_t* vr = &sVT[(h * HD + nt * 16 + lr) * VTS + ks * 32];
        bv.u[0] = *(const uint4*)(vr + lh * 8);
        bv.u[1] = *(const uint4*)(vr + 16 + lh * 8);
        o[nt].v = __builtin_amdgcn_wmma_f32_16x16x32_bf16(
            false, ap.v, false, bv.v, (short)0, o[nt].v, false, false);
      }
    }

    // store O tile into AO (= sX) bf16 row-major
#pragma unroll
    for (int nt = 0; nt < 2; ++nt) {
#pragma unroll
      for (int i = 0; i < 8; ++i) {
        int m = mt * 16 + i + lh * 8;
        int n = h * HD + nt * 16 + lr;
        sX[m * XS + n] = (bf16_t)o[nt].f[i];
      }
    }
  }
  __syncthreads();

  // ---------------- proj GEMM: AO[64,512] @ Wp^T + b ----------------
  float* og = out + (size_t)win * NTOK * DIM;
  {
    const int oc0 = wave * 32 + lr;
    const int oc1 = wave * 32 + 16 + lr;
    const bf16_t* wrow0 = proj_wb + (size_t)oc0 * DIM;
    const bf16_t* wrow1 = proj_wb + (size_t)oc1 * DIM;
    AccF acc0[4], acc1[4];
#pragma unroll
    for (int m = 0; m < 4; ++m) { acc0[m].v = zero8(); acc1[m].v = zero8(); }

    for (int kk = 0; kk < DIM; kk += 32) {
      __builtin_prefetch(wrow0 + kk + 128, 0, 1);
      __builtin_prefetch(wrow1 + kk + 128, 0, 1);
      FragBF bf0, bf1;
      bf0.u[0] = *(const uint4*)(wrow0 + kk + lh * 8);
      bf0.u[1] = *(const uint4*)(wrow0 + kk + 16 + lh * 8);
      bf1.u[0] = *(const uint4*)(wrow1 + kk + lh * 8);
      bf1.u[1] = *(const uint4*)(wrow1 + kk + 16 + lh * 8);
#pragma unroll
      for (int m = 0; m < 4; ++m) {
        FragBF afrag;
        const bf16_t* ar = &sX[(m * 16 + lr) * XS + kk];
        afrag.u[0] = *(const uint4*)(ar + lh * 8);
        afrag.u[1] = *(const uint4*)(ar + 16 + lh * 8);
        acc0[m].v = __builtin_amdgcn_wmma_f32_16x16x32_bf16(
            false, afrag.v, false, bf0.v, (short)0, acc0[m].v, false, false);
        acc1[m].v = __builtin_amdgcn_wmma_f32_16x16x32_bf16(
            false, afrag.v, false, bf1.v, (short)0, acc1[m].v, false, false);
      }
    }
    const float b0 = proj_b[oc0];
    const float b1 = proj_b[oc1];
#pragma unroll
    for (int m = 0; m < 4; ++m) {
#pragma unroll
      for (int i = 0; i < 8; ++i) {
        int row = m * 16 + i + lh * 8;
        og[row * DIM + oc0] = acc0[m].f[i] + b0;
        og[row * DIM + oc1] = acc1[m].f[i] + b1;
      }
    }
  }
}

// ---------------------------------------------------------------
extern "C" void kernel_launch(void* const* d_in, const int* in_sizes, int n_in,
                              void* d_out, int out_size, void* d_ws, size_t ws_size,
                              hipStream_t stream) {
  const float* x          = (const float*)d_in[0];
  const float* mask       = (const float*)d_in[1];
  const float* qkv_w      = (const float*)d_in[2];
  const float* qkv_b      = (const float*)d_in[3];
  const float* proj_w     = (const float*)d_in[4];
  const float* proj_b     = (const float*)d_in[5];
  const float* bias_table = (const float*)d_in[6];
  const int*   rel_idx    = (const int*)d_in[7];
  float* out = (float*)d_out;

  char* ws = (char*)d_ws;
  const size_t qkvw_bytes = (size_t)3 * DIM * DIM * sizeof(bf16_t); // 1.5 MB
  const size_t pw_bytes   = (size_t)DIM * DIM * sizeof(bf16_t);     // 0.5 MB
  bf16_t* qkv_wb   = (bf16_t*)ws;
  bf16_t* proj_wb  = (bf16_t*)(ws + qkvw_bytes);
  float*  bias_exp = (float*)(ws + qkvw_bytes + pw_bytes);          // 256 KB

  const int prep_elems = 3 * DIM * DIM;
  winattn_prep<<<(prep_elems + 255) / 256, 256, 0, stream>>>(
      qkv_w, proj_w, bias_table, rel_idx, qkv_wb, proj_wb, bias_exp);

  winattn_main<<<NWINDOWS, THREADS, 0, stream>>>(
      x, mask, qkv_wb, qkv_b, proj_wb, proj_b, bias_exp, out);
}
